// GraphGDP_GraphGPS_79766132621913
// MI455X (gfx1250) — matrix operand order repaired and checked by reading, not compiled, not run on hardware
//
#include <hip/hip_runtime.h>
#include <hip/hip_bf16.h>
#include <math.h>

// ---------------- problem constants ----------------
#define NNODES 50000
#define NEDGES 800000
#define CDIM   128
#define PEDIM  12
#define NODE_IN 422
#define HEADS  4
#define DH     32
#define MFEAT  110   // performer random features
#define MP     128   // padded feature dim
#define NG     50    // graphs
#define SG     1000  // nodes per graph
#define SP     1024  // padded per-graph sequence (K for ctx gemm)
#define NLAYERS 4

typedef __attribute__((ext_vector_type(16))) __bf16 bf16x16;
typedef __attribute__((ext_vector_type(8)))  __bf16 bf16x8;
typedef __attribute__((ext_vector_type(8)))  float  f32x8;

// ---------------- small device helpers ----------------
__device__ __forceinline__ __bf16 f2b(float f) {
  unsigned u = __builtin_bit_cast(unsigned, f);
  unsigned r = 0x7FFFu + ((u >> 16) & 1u);
  unsigned short s = (unsigned short)((u + r) >> 16);
  return __builtin_bit_cast(__bf16, s);
}
__device__ __forceinline__ float b2f(__bf16 b) {
  unsigned short s = __builtin_bit_cast(unsigned short, b);
  unsigned u = ((unsigned)s) << 16;
  return __builtin_bit_cast(float, u);
}
__device__ __forceinline__ float actf(float x, int act) {
  if (act == 1) return fmaxf(x, 0.f);
  if (act == 2) return 0.5f * x * (1.f + erff(x * 0.70710678118654752f)); // exact gelu
  return x;
}

// ---------------- staging conversions ----------------
// Xb[m*Kp+k] = bf16(X[m*ldx + col0 + k] * scale), zero-padded k>=K
__global__ __launch_bounds__(256) void conv_bf16(const float* __restrict__ X, int ldx, int col0,
                                                 int K, int Kp, long M, float scale,
                                                 __bf16* __restrict__ Xb) {
  long idx = (long)blockIdx.x * 256 + threadIdx.x;
  long total = M * (long)Kp;
  if (idx >= total) return;
  long m = idx / Kp; int k = (int)(idx % Kp);
  float v = (k < K) ? X[m * (long)ldx + col0 + k] * scale : 0.f;
  Xb[idx] = f2b(v);
}

// WTb[n*Kp+k] = bf16(W[k*Nact+n]) (or W[n*K+k] if pre-transposed), zero-padded
__global__ __launch_bounds__(256) void conv_wt(const float* __restrict__ W, int K, int Nact,
                                               int Kp, int Npad, int pret,
                                               __bf16* __restrict__ WTb) {
  long idx = (long)blockIdx.x * 256 + threadIdx.x;
  long total = (long)Npad * Kp;
  if (idx >= total) return;
  int n = (int)(idx / Kp), k = (int)(idx % Kp);
  float v = 0.f;
  if (n < Nact && k < K) v = pret ? W[(long)n * K + k] : W[(long)k * Nact + n];
  WTb[idx] = f2b(v);
}

// ---------------- generic bf16 WMMA GEMM ----------------
// Y[M,ldy] = act( Xb[M,Kp] * WTb[Npad,Kp]^T + bias + res ), cols < nact stored
__global__ __launch_bounds__(256) void wmma_gemm(const __bf16* __restrict__ Xb,
                                                 const __bf16* __restrict__ WTb,
                                                 const float* __restrict__ bias,
                                                 const float* __restrict__ res,
                                                 float* __restrict__ Y,
                                                 int M, int nact, int Kp, int ldy, int act) {
  int w = threadIdx.x >> 5, lane = threadIdx.x & 31;
  int hs = lane >> 4, ln = lane & 15;
  int rowT = blockIdx.x * 64 + (w & 3) * 16;
  int colBase = blockIdx.y * 64 + (w >> 2) * 32;
  int arow = rowT + ln; if (arow >= M) arow = M - 1;
  const __bf16* ap  = Xb  + (long)arow * Kp;
  const __bf16* bp0 = WTb + (long)(colBase + ln) * Kp;
  const __bf16* bp1 = WTb + (long)(colBase + 16 + ln) * Kp;
  f32x8 c0 = {0,0,0,0,0,0,0,0}, c1 = {0,0,0,0,0,0,0,0};
  int nkb = Kp >> 5;
  for (int kb = 0; kb < nkb; ++kb) {
    int koA = kb * 32 + hs * 16;
    int koB = kb * 32 + hs * 16;
    bf16x8 a0 = *(const bf16x8*)(ap + koA);
    bf16x8 a1 = *(const bf16x8*)(ap + koA + 8);
    bf16x16 a, b0, b1;
    b0 = *(const bf16x16*)(bp0 + koB);
    b1 = *(const bf16x16*)(bp1 + koB);
#pragma unroll
    for (int i = 0; i < 8; i++) { a[i] = a0[i]; a[i + 8] = a1[i]; }
    c0 = __builtin_amdgcn_wmma_f32_16x16x32_bf16(false, a, false, b0, (short)0, c0, false, false);
    c1 = __builtin_amdgcn_wmma_f32_16x16x32_bf16(false, a, false, b1, (short)0, c1, false, false);
  }
#pragma unroll
  for (int r = 0; r < 8; r++) {
    int orow = rowT + hs * 8 + r;
    if (orow >= M) continue;
    int cc0 = colBase + ln, cc1 = colBase + 16 + ln;
    if (cc0 < nact) {
      float s = c0[r];
      if (bias) s += bias[cc0];
      if (res)  s += res[(long)orow * ldy + cc0];
      Y[(long)orow * ldy + cc0] = actf(s, act);
    }
    if (cc1 < nact) {
      float s = c1[r];
      if (bias) s += bias[cc1];
      if (res)  s += res[(long)orow * ldy + cc1];
      Y[(long)orow * ldy + cc1] = actf(s, act);
    }
  }
}

// ---------------- performer specialized GEMMs ----------------
// ctxT[hg][d][m] = sum_s kpT[hg][m][s] * vT[hg][d][s]   (K = SP = 1024)
__global__ __launch_bounds__(256) void ctx_gemm(const __bf16* __restrict__ kpT,
                                                const __bf16* __restrict__ vT,
                                                __bf16* __restrict__ ctxT) {
  int hg = blockIdx.x;
  int w = threadIdx.x >> 5, lane = threadIdx.x & 31;
  int hs = lane >> 4, ln = lane & 15;
  const __bf16* ap  = kpT + ((long)hg * MP + w * 16 + ln) * SP;
  const __bf16* bp0 = vT  + ((long)hg * DH + ln) * SP;
  const __bf16* bp1 = vT  + ((long)hg * DH + 16 + ln) * SP;
  f32x8 c0 = {0,0,0,0,0,0,0,0}, c1 = {0,0,0,0,0,0,0,0};
  for (int kb = 0; kb < SP / 32; ++kb) {
    int koA = kb * 32 + hs * 16;
    int koB = kb * 32 + hs * 16;
    bf16x8 a0 = *(const bf16x8*)(ap + koA);
    bf16x8 a1 = *(const bf16x8*)(ap + koA + 8);
    bf16x16 a, b0, b1;
    b0 = *(const bf16x16*)(bp0 + koB);
    b1 = *(const bf16x16*)(bp1 + koB);
#pragma unroll
    for (int i = 0; i < 8; i++) { a[i] = a0[i]; a[i + 8] = a1[i]; }
    c0 = __builtin_amdgcn_wmma_f32_16x16x32_bf16(false, a, false, b0, (short)0, c0, false, false);
    c1 = __builtin_amdgcn_wmma_f32_16x16x32_bf16(false, a, false, b1, (short)0, c1, false, false);
  }
#pragma unroll
  for (int r = 0; r < 8; r++) {
    int m = w * 16 + hs * 8 + r;
    ctxT[((long)hg * DH + ln) * MP + m]      = f2b(c0[r]);
    ctxT[((long)hg * DH + 16 + ln) * MP + m] = f2b(c1[r]);
  }
}

// O[n][h*32+d] = (sum_m qp[h][n][m] * ctxT[h,g][d][m]) / Dn[h][n]
__global__ __launch_bounds__(256) void o_gemm(const __bf16* __restrict__ QP,
                                              const __bf16* __restrict__ ctxT,
                                              const float* __restrict__ Dn,
                                              float* __restrict__ O) {
  int h = blockIdx.y;
  int g = blockIdx.x >> 4, t = blockIdx.x & 15;
  int rowBase = g * SG + t * 64, rowEnd = g * SG + SG;
  int w = threadIdx.x >> 5, lane = threadIdx.x & 31;
  int hs = lane >> 4, ln = lane & 15;
  int mt = w & 3, ngp = w >> 2;
  int arow = rowBase + mt * 16 + ln; if (arow >= rowEnd) arow = rowEnd - 1;
  const __bf16* ap = QP + ((long)h * NNODES + arow) * MP;
  const __bf16* bp = ctxT + ((long)(h * NG + g) * DH + ngp * 16 + ln) * MP;
  f32x8 c = {0,0,0,0,0,0,0,0};
  for (int kb = 0; kb < MP / 32; ++kb) {
    int koA = kb * 32 + hs * 16;
    int koB = kb * 32 + hs * 16;
    bf16x8 a0 = *(const bf16x8*)(ap + koA);
    bf16x8 a1 = *(const bf16x8*)(ap + koA + 8);
    bf16x16 a, b;
    b = *(const bf16x16*)(bp + koB);
#pragma unroll
    for (int i = 0; i < 8; i++) { a[i] = a0[i]; a[i + 8] = a1[i]; }
    c = __builtin_amdgcn_wmma_f32_16x16x32_bf16(false, a, false, b, (short)0, c, false, false);
  }
#pragma unroll
  for (int r = 0; r < 8; r++) {
    int orow = rowBase + mt * 16 + hs * 8 + r;
    if (orow < rowEnd) {
      int d = ngp * 16 + ln;
      O[(long)orow * CDIM + h * DH + d] = c[r] / Dn[(long)h * NNODES + orow];
    }
  }
}

// ---------------- elementwise / graph kernels ----------------
__global__ __launch_bounds__(256) void fill_u32(unsigned* __restrict__ p, long n, unsigned v) {
  long i = (long)blockIdx.x * 256 + threadIdx.x;
  if (i < n) p[i] = v;
}
__global__ __launch_bounds__(256) void add_k(const float* __restrict__ A, const float* __restrict__ B,
                                             float* __restrict__ Y, long n) {
  long i = (long)blockIdx.x * 256 + threadIdx.x;
  if (i < n) Y[i] = A[i] + B[i];
}
// GINE message + scatter-add: aggr[dst] += relu(h[src] + eattr*We + be)
__global__ __launch_bounds__(128) void gine_aggr(const float* __restrict__ h,
                                                 const int* __restrict__ src, const int* __restrict__ dst,
                                                 const float* __restrict__ eattr,
                                                 const float* __restrict__ We, const float* __restrict__ be,
                                                 float* __restrict__ aggr) {
  int e = blockIdx.x, c = threadIdx.x;
  float m = h[(long)src[e] * CDIM + c] + eattr[e] * We[c] + be[c];
  m = fmaxf(m, 0.f);
  atomicAdd(&aggr[(long)dst[e] * CDIM + c], m);
}
__global__ __launch_bounds__(256) void bn_act(const float* __restrict__ X, const float* __restrict__ R,
                                              const float* __restrict__ g, const float* __restrict__ b,
                                              const float* __restrict__ rm, const float* __restrict__ rv,
                                              float* __restrict__ Y, long rows, int Cc, int act) {
  long i = (long)blockIdx.x * 256 + threadIdx.x;
  if (i >= rows * (long)Cc) return;
  int c = (int)(i % Cc);
  float x = X[i]; if (R) x += R[i];
  float y = (x - rm[c]) * rsqrtf(rv[c] + 1e-5f) * g[c] + b[c];
  Y[i] = actf(y, act);
}
__global__ __launch_bounds__(128) void graph_stats_k(const float* __restrict__ h,
                                                     float* __restrict__ mu, float* __restrict__ rsd) {
  int g = blockIdx.x, c = threadIdx.x;
  float s = 0.f, s2 = 0.f;
  for (int i = 0; i < SG; i++) {
    float v = h[((long)g * SG + i) * CDIM + c];
    s += v; s2 += v * v;
  }
  float m = s / SG;
  float var = (s2 - SG * m * m) / (SG - 1);
  mu[g * CDIM + c] = m;
  rsd[g * CDIM + c] = 1.f / (sqrtf(fmaxf(var, 0.f)) + 1e-6f);
}
__global__ __launch_bounds__(256) void standardize_k(const float* __restrict__ h,
                                                     const float* __restrict__ mu, const float* __restrict__ rsd,
                                                     float* __restrict__ z) {
  long i = (long)blockIdx.x * 256 + threadIdx.x;
  if (i >= (long)NNODES * CDIM) return;
  int row = (int)(i / CDIM), c = (int)(i % CDIM);
  int g = row / SG;
  z[i] = (h[i] - mu[g * CDIM + c]) * rsd[g * CDIM + c];
}
__global__ __launch_bounds__(128) void layernorm_k(const float* __restrict__ X,
                                                   const float* __restrict__ g, const float* __restrict__ b,
                                                   float* __restrict__ Y) {
  __shared__ float red[128];
  int n = blockIdx.x, c = threadIdx.x;
  float x = X[(long)n * CDIM + c];
  red[c] = x; __syncthreads();
  for (int s = 64; s > 0; s >>= 1) { if (c < s) red[c] += red[c + s]; __syncthreads(); }
  float mu = red[0] / CDIM; __syncthreads();
  float d = x - mu; red[c] = d * d; __syncthreads();
  for (int s = 64; s > 0; s >>= 1) { if (c < s) red[c] += red[c + s]; __syncthreads(); }
  float var = red[0] / CDIM;
  Y[(long)n * CDIM + c] = d * rsqrtf(var + 1e-5f) * g[c] + b[c];
}
// qp = ratio*(exp(dd - diag - rowmax) + 1e-4), m<110 ; stored bf16 [H][N][MP]
__global__ __launch_bounds__(128) void make_qp_k(const float* __restrict__ DD, const float* __restrict__ Q,
                                                 __bf16* __restrict__ QP, float dn, float ratio) {
  __shared__ float red[128];
  __shared__ float sdiag;
  int n = blockIdx.x, h = blockIdx.y, m = threadIdx.x;
  long base = ((long)h * NNODES + n) * MP;
  float dd = DD[base + m];
  bool valid = m < MFEAT;
  red[m] = valid ? dd : -3.4e38f; __syncthreads();
  for (int s = 64; s > 0; s >>= 1) { if (m < s) red[m] = fmaxf(red[m], red[m + s]); __syncthreads(); }
  float rmax = red[0];
  if (m == 0) {
    float s = 0.f; const float* q = Q + (long)n * CDIM + h * DH;
    for (int d = 0; d < DH; d++) { float v = q[d] * dn; s += v * v; }
    sdiag = 0.5f * s;
  }
  __syncthreads();
  float val = valid ? ratio * (expf(dd - sdiag - rmax) + 1e-4f) : 0.f;
  QP[base + m] = f2b(val);
}
__global__ __launch_bounds__(256) void kmax_k(const float* __restrict__ DD, float* __restrict__ kmax) {
  __shared__ float red[256];
  int g = blockIdx.x, h = blockIdx.y, t = threadIdx.x;
  float m = -3.4e38f;
  long base = ((long)h * NNODES + (long)g * SG) * MP;
  for (int i = t; i < SG * MFEAT; i += 256) {
    int s = i / MFEAT, mm = i - s * MFEAT;
    m = fmaxf(m, DD[base + (long)s * MP + mm]);
  }
  red[t] = m; __syncthreads();
  for (int s = 128; s > 0; s >>= 1) { if (t < s) red[t] = fmaxf(red[t], red[t + s]); __syncthreads(); }
  if (t == 0) kmax[h * NG + g] = red[0];
}
// kp written transposed: kpT[h,g][m][s_local] (pads stay zero-filled)
__global__ __launch_bounds__(128) void make_kp_k(const float* __restrict__ DD, const float* __restrict__ K,
                                                 const float* __restrict__ kmax, __bf16* __restrict__ kpT,
                                                 float dn, float ratio) {
  __shared__ float sdiag;
  int n = blockIdx.x, h = blockIdx.y, m = threadIdx.x;
  long base = ((long)h * NNODES + n) * MP;
  float dd = DD[base + m];
  if (m == 0) {
    float s = 0.f; const float* k = K + (long)n * CDIM + h * DH;
    for (int d = 0; d < DH; d++) { float v = k[d] * dn; s += v * v; }
    sdiag = 0.5f * s;
  }
  __syncthreads();
  int g = n / SG, sl = n - g * SG;
  if (m < MFEAT) {
    float val = ratio * (expf(dd - sdiag - kmax[h * NG + g]) + 1e-4f);
    kpT[(((long)h * NG + g) * MP + m) * SP + sl] = f2b(val);
  }
}
__global__ __launch_bounds__(128) void make_vT_k(const float* __restrict__ V, __bf16* __restrict__ vT) {
  int n = blockIdx.x, t = threadIdx.x;
  int h = t >> 5, d = t & 31;
  int g = n / SG, sl = n - g * SG;
  vT[(((long)h * NG + g) * DH + d) * SP + sl] = f2b(V[(long)n * CDIM + h * DH + d]);
}
__global__ __launch_bounds__(128) void kpsum_k(const __bf16* __restrict__ kpT, float* __restrict__ kpsum) {
  int g = blockIdx.x, h = blockIdx.y, m = threadIdx.x;
  long base = (((long)h * NG + g) * MP + m) * SP;
  float s = 0.f;
  for (int i = 0; i < SG; i++) s += b2f(kpT[base + i]);
  kpsum[((long)h * NG + g) * MP + m] = s;
}
__global__ __launch_bounds__(128) void dnorm_k(const __bf16* __restrict__ QP, const float* __restrict__ kpsum,
                                               float* __restrict__ Dn) {
  __shared__ float red[128];
  int n = blockIdx.x, h = blockIdx.y, m = threadIdx.x;
  int g = n / SG;
  float p = b2f(QP[((long)h * NNODES + n) * MP + m]) * kpsum[((long)h * NG + g) * MP + m];
  red[m] = p; __syncthreads();
  for (int s = 64; s > 0; s >>= 1) { if (m < s) red[m] += red[m + s]; __syncthreads(); }
  if (m == 0) Dn[(long)h * NNODES + n] = red[0];
}
__global__ __launch_bounds__(256) void absmax_k(const float* __restrict__ X, long n, unsigned* __restrict__ mx) {
  __shared__ float red[256];
  float m = 0.f;
  for (long i = (long)blockIdx.x * 256 + threadIdx.x; i < n; i += (long)gridDim.x * 256)
    m = fmaxf(m, fabsf(X[i]));
  red[threadIdx.x] = m; __syncthreads();
  for (int s = 128; s > 0; s >>= 1) { if (threadIdx.x < s) red[threadIdx.x] = fmaxf(red[threadIdx.x], red[threadIdx.x + s]); __syncthreads(); }
  if (threadIdx.x == 0) atomicMax(mx, __float_as_uint(red[0]));
}
__global__ __launch_bounds__(256) void clamp_k(const float* __restrict__ X, float* __restrict__ Y,
                                               long n, float t, const unsigned* __restrict__ mx) {
  long i = (long)blockIdx.x * 256 + threadIdx.x;
  if (i >= n) return;
  float m = __uint_as_float(*mx);
  float x = X[i];
  Y[i] = (m > t) ? tanhf(x / t) * t : x;
}
__global__ __launch_bounds__(256) void pe_k(const float* __restrict__ lap,
                                            const float* __restrict__ g, const float* __restrict__ b,
                                            const float* __restrict__ rm, const float* __restrict__ rv,
                                            const float* __restrict__ Wl, const float* __restrict__ bl,
                                            float* __restrict__ h) {
  long i = (long)blockIdx.x * 256 + threadIdx.x;
  if (i >= (long)NNODES * PEDIM) return;
  int n = (int)(i / PEDIM), j = (int)(i % PEDIM);
  float acc = bl[j];
  for (int k = 0; k < PEDIM; k++) {
    float t = (lap[(long)n * PEDIM + k] - rm[k]) * rsqrtf(rv[k] + 1e-5f) * g[k] + b[k];
    acc += t * Wl[k * PEDIM + j];
  }
  h[(long)n * CDIM + (CDIM - PEDIM) + j] = acc;
}
__global__ __launch_bounds__(128) void pool_sum_k(const float* __restrict__ h, float* __restrict__ gf) {
  int g = blockIdx.x, c = threadIdx.x;
  float s = 0.f;
  for (int i = 0; i < SG; i++) s += h[((long)g * SG + i) * CDIM + c];
  gf[g * CDIM + c] = s;
}

// ---------------- host orchestration ----------------
extern "C" void kernel_launch(void* const* d_in, const int* in_sizes, int n_in,
                              void* d_out, int out_size, void* d_ws, size_t ws_size,
                              hipStream_t stream) {
  (void)in_sizes; (void)n_in; (void)out_size; (void)ws_size;
  auto P = [&](int i) -> const float* { return (const float*)d_in[i]; };
  const float* x     = P(0);
  const int*   eidx  = (const int*)d_in[1];
  const float* eattr = P(2);
  const float* lap   = P(4);
  const int* esrc = eidx;
  const int* edst = eidx + NEDGES;
  const int LBASE = 15, LSTRIDE = 37;

  // -------- workspace carve-up --------
  char* wsp = (char*)d_ws; size_t off = 0;
  auto alloc = [&](size_t bytes) -> void* {
    void* p = wsp + off; off += (bytes + 255) & ~(size_t)255; return p;
  };
  const long NC = (long)NNODES * CDIM;
  float* h    = (float*)alloc(NC * 4);
  float* aggr = (float*)alloc(NC * 4);
  float* hb1  = (float*)alloc(NC * 4);
  float* z    = (float*)alloc(NC * 4);
  float* z1   = (float*)alloc(NC * 4);
  float* bufA = (float*)alloc(NC * 4);
  float* bufB = (float*)alloc(NC * 4);
  float* bufC = (float*)alloc(NC * 4);
  float* bufD = (float*)alloc(NC * 4);
  float* t512 = (float*)alloc((size_t)NNODES * 512 * 4); // wide tmp; also dd[H][N][MP]
  float* mu    = (float*)alloc(NG * CDIM * 4);
  float* rsd   = (float*)alloc(NG * CDIM * 4);
  float* kmax  = (float*)alloc(HEADS * NG * 4);
  float* kpsum = (float*)alloc((size_t)HEADS * NG * MP * 4);
  float* Dn    = (float*)alloc((size_t)HEADS * NNODES * 4);
  float* gf    = (float*)alloc(NG * CDIM * 4);
  float* sg1   = (float*)alloc(NG * 64 * 4);
  float* sg2   = (float*)alloc(NG * 64 * 4);
  float* sg3   = (float*)alloc(NG * 2 * 4);
  unsigned* mx = (unsigned*)alloc(256);
  __bf16* Xb   = (__bf16*)alloc((size_t)NNODES * 512 * 2);
  __bf16* QP   = (__bf16*)alloc((size_t)HEADS * NNODES * MP * 2);
  __bf16* kpT  = (__bf16*)alloc((size_t)HEADS * NG * MP * SP * 2);
  __bf16* vT   = (__bf16*)alloc((size_t)HEADS * NG * DH * SP * 2);
  __bf16* ctxT = (__bf16*)alloc((size_t)HEADS * NG * DH * MP * 2);
  __bf16* wpool = (__bf16*)alloc((size_t)(2u << 20) * 2);
  size_t woff = 0;

  auto cwt = [&](const float* W, int K, int Nact, int Kp, int Npad, int pret) -> __bf16* {
    __bf16* dst = wpool + woff; woff += (size_t)Npad * Kp;
    long tot = (long)Npad * Kp;
    conv_wt<<<dim3((unsigned)((tot + 255) / 256)), 256, 0, stream>>>(W, K, Nact, Kp, Npad, pret, dst);
    return dst;
  };
  auto conv = [&](const float* X, int ldx, int col0, int K, int Kp, long M, float scale) {
    long tot = M * (long)Kp;
    conv_bf16<<<dim3((unsigned)((tot + 255) / 256)), 256, 0, stream>>>(X, ldx, col0, K, Kp, M, scale, Xb);
  };
  auto gemm = [&](const __bf16* wt, const float* bias, const float* res, float* Y,
                  long M, int nact, int Kp, int ldy, int act) {
    dim3 gd((unsigned)((M + 63) / 64), (unsigned)((nact + 63) / 64));
    wmma_gemm<<<gd, 256, 0, stream>>>(Xb, wt, bias, res, Y, (int)M, nact, Kp, ldy, act);
  };
  auto fillz = [&](void* p, size_t bytes) {
    long n = (long)(bytes / 4);
    fill_u32<<<dim3((unsigned)((n + 255) / 256)), 256, 0, stream>>>((unsigned*)p, n, 0u);
  };
  auto absmax = [&](const float* X, long n) {
    fill_u32<<<1, 256, 0, stream>>>(mx, 1, 0u);
    long blocks = (n + 255) / 256; if (blocks > 4096) blocks = 4096;
    absmax_k<<<dim3((unsigned)blocks), 256, 0, stream>>>(X, n, mx);
  };
  auto clampk = [&](const float* X, float* Y, long n, float t) {
    clamp_k<<<dim3((unsigned)((n + 255) / 256)), 256, 0, stream>>>(X, Y, n, t, mx);
  };
  auto addk = [&](const float* A, const float* B, float* Y, long n) {
    add_k<<<dim3((unsigned)((n + 255) / 256)), 256, 0, stream>>>(A, B, Y, n);
  };
  auto bnk = [&](const float* X, const float* R, int pidx, float* Y, long rows, int Cc, int act) {
    bn_act<<<dim3((unsigned)((rows * Cc + 255) / 256)), 256, 0, stream>>>(
        X, R, P(pidx), P(pidx + 1), P(pidx + 2), P(pidx + 3), Y, rows, Cc, act);
  };

  // -------- stage all weights (bf16, transposed, padded) --------
  __bf16* wnode = cwt(P(5), NODE_IN, CDIM - PEDIM, 448, 128, 0);
  __bf16 *wnn1[NLAYERS], *wnn2[NLAYERS], *wq[NLAYERS], *wk[NLAYERS], *wv[NLAYERS], *wo[NLAYERS],
         *wm1[NLAYERS], *wm2[NLAYERS], *wf1[NLAYERS], *wf2[NLAYERS], *wpj[NLAYERS];
  for (int l = 0; l < NLAYERS; l++) {
    int B = LBASE + LSTRIDE * l;
    wnn1[l] = cwt(P(B + 0), 128, 128, 128, 128, 0);
    wnn2[l] = cwt(P(B + 2), 128, 128, 128, 128, 0);
    wm1[l]  = cwt(P(B + 16), 128, 256, 128, 256, 0);
    wm2[l]  = cwt(P(B + 18), 256, 128, 256, 128, 0);
    wq[l]   = cwt(P(B + 22), 128, 128, 128, 128, 0);
    wk[l]   = cwt(P(B + 24), 128, 128, 128, 128, 0);
    wv[l]   = cwt(P(B + 26), 128, 128, 128, 128, 0);
    wo[l]   = cwt(P(B + 28), 128, 128, 128, 128, 0);
    wpj[l]  = cwt(P(B + 30), DH, MFEAT, DH, MP, 1); // proj [110,32] already row-per-output
    wf1[l]  = cwt(P(B + 33), 128, 512, 128, 512, 0);
    wf2[l]  = cwt(P(B + 35), 512, 128, 512, 128, 0);
  }
  __bf16* wnh1 = cwt(P(163), 128, 64, 128, 64, 0);
  __bf16* wnh2 = cwt(P(169), 64, 2, 64, 64, 0);
  __bf16* wgh1 = cwt(P(171), 128, 64, 128, 64, 0);
  __bf16* wgh2 = cwt(P(177), 64, 2, 64, 64, 0);

  // zero padded sequence buffers once (pads never written afterwards)
  fillz(kpT, (size_t)HEADS * NG * MP * SP * 2);
  fillz(vT, (size_t)HEADS * NG * DH * SP * 2);

  // -------- embedding --------
  conv(x, NODE_IN, 0, NODE_IN, 448, NNODES, 1.f);
  gemm(wnode, P(6), nullptr, h, NNODES, CDIM - PEDIM, 448, CDIM, 0);
  pe_k<<<dim3((unsigned)(((long)NNODES * PEDIM + 255) / 256)), 256, 0, stream>>>(
      lap, P(7), P(8), P(9), P(10), P(11), P(12), h);

  const float dn = 1.0f / sqrtf(sqrtf((float)DH));   // DH^-0.25
  const float ratio = 1.0f / sqrtf((float)MFEAT);

  // -------- GPS layers --------
  for (int l = 0; l < NLAYERS; l++) {
    int B = LBASE + LSTRIDE * l;
    // GINEConv branch
    fillz(aggr, NC * 4);
    gine_aggr<<<dim3(NEDGES), 128, 0, stream>>>(h, esrc, edst, eattr, P(13), P(14), aggr);
    addk(h, aggr, bufA, NC);
    conv(bufA, CDIM, 0, CDIM, CDIM, NNODES, 1.f);
    gemm(wnn1[l], P(B + 1), nullptr, bufA, NNODES, CDIM, CDIM, CDIM, 1);
    conv(bufA, CDIM, 0, CDIM, CDIM, NNODES, 1.f);
    gemm(wnn2[l], P(B + 3), nullptr, bufB, NNODES, CDIM, CDIM, CDIM, 0);
    bnk(bufB, h, B + 4, hb1, NNODES, CDIM, 0);          // norm1(h_lin + x)
    // per-graph standardize of layer input
    graph_stats_k<<<dim3(NG), 128, 0, stream>>>(h, mu, rsd);
    standardize_k<<<dim3((unsigned)((NC + 255) / 256)), 256, 0, stream>>>(h, mu, rsd, z);
    // performer attention
    layernorm_k<<<dim3(NNODES), 128, 0, stream>>>(z, P(B + 20), P(B + 21), bufA);
    conv(bufA, CDIM, 0, CDIM, CDIM, NNODES, 1.f);
    gemm(wq[l], nullptr, nullptr, bufB, NNODES, CDIM, CDIM, CDIM, 0);
    gemm(wk[l], nullptr, nullptr, bufC, NNODES, CDIM, CDIM, CDIM, 0);
    gemm(wv[l], nullptr, nullptr, bufD, NNODES, CDIM, CDIM, CDIM, 0);
    for (int hh = 0; hh < HEADS; hh++) {                // dd_q = (q*dn) @ proj^T
      conv(bufB, CDIM, hh * DH, DH, DH, NNODES, dn);
      gemm(wpj[l], nullptr, nullptr, t512 + (size_t)hh * NNODES * MP, NNODES, MP, DH, MP, 0);
    }
    make_qp_k<<<dim3(NNODES, HEADS), 128, 0, stream>>>(t512, bufB, QP, dn, ratio);
    for (int hh = 0; hh < HEADS; hh++) {                // dd_k
      conv(bufC, CDIM, hh * DH, DH, DH, NNODES, dn);
      gemm(wpj[l], nullptr, nullptr, t512 + (size_t)hh * NNODES * MP, NNODES, MP, DH, MP, 0);
    }
    kmax_k<<<dim3(NG, HEADS), 256, 0, stream>>>(t512, kmax);
    make_kp_k<<<dim3(NNODES, HEADS), 128, 0, stream>>>(t512, bufC, kmax, kpT, dn, ratio);
    make_vT_k<<<dim3(NNODES), 128, 0, stream>>>(bufD, vT);
    kpsum_k<<<dim3(NG, HEADS), 128, 0, stream>>>(kpT, kpsum);
    ctx_gemm<<<dim3(HEADS * NG), 256, 0, stream>>>(kpT, vT, ctxT);
    dnorm_k<<<dim3(NNODES, HEADS), 128, 0, stream>>>(QP, kpsum, Dn);
    o_gemm<<<dim3(NG * 16, HEADS), 256, 0, stream>>>(QP, ctxT, Dn, bufA);
    conv(bufA, CDIM, 0, CDIM, CDIM, NNODES, 1.f);
    gemm(wo[l], P(B + 29), z, z1, NNODES, CDIM, CDIM, CDIM, 0);  // z1 = z + Wo(o)
    layernorm_k<<<dim3(NNODES), 128, 0, stream>>>(z1, P(B + 31), P(B + 32), bufB);
    conv(bufB, CDIM, 0, CDIM, CDIM, NNODES, 1.f);
    gemm(wf1[l], P(B + 34), nullptr, t512, NNODES, 512, CDIM, 512, 2);  // gelu
    conv(t512, 512, 0, 512, 512, NNODES, 1.f);
    gemm(wf2[l], P(B + 36), z1, bufC, NNODES, CDIM, 512, CDIM, 0);      // z2 = z1 + ff2(...)
    // ha = bn2(soft_clamp(z2 + h, 100))
    addk(bufC, h, bufD, NC);
    absmax(bufD, NC); clampk(bufD, bufD, NC, 100.f);
    bnk(bufD, nullptr, B + 8, bufD, NNODES, CDIM, 0);
    // out = bn3(soft_clamp(hb1 + ha, 1000))
    addk(hb1, bufD, bufC, NC);
    absmax(bufC, NC); clampk(bufC, bufC, NC, 1000.f);
    bnk(bufC, nullptr, B + 12, bufC, NNODES, CDIM, 0);
    // mlp
    conv(bufC, CDIM, 0, CDIM, CDIM, NNODES, 1.f);
    gemm(wm1[l], P(B + 17), nullptr, t512, NNODES, 256, CDIM, 256, 1);
    conv(t512, 256, 0, 256, 256, NNODES, 1.f);
    gemm(wm2[l], P(B + 19), nullptr, bufA, NNODES, CDIM, 256, CDIM, 0);
    addk(bufA, bufC, bufB, NC);
    absmax(bufB, NC); clampk(bufB, bufB, NC, 1000.f);   // inner soft_clamp
    absmax(bufB, NC); clampk(bufB, h, NC, 1000.f);      // outer soft_clamp -> new h
  }

  // -------- heads --------
  conv(h, CDIM, 0, CDIM, CDIM, NNODES, 1.f);
  gemm(wnh1, P(164), nullptr, bufA, NNODES, 64, CDIM, 64, 0);
  bnk(bufA, nullptr, 165, bufB, NNODES, 64, 2);         // gelu(bn(...))
  conv(bufB, 64, 0, 64, 64, NNODES, 1.f);
  gemm(wnh2, P(170), nullptr, bufC, NNODES, 2, 64, 2, 0);
  absmax(bufC, (long)NNODES * 2);
  clampk(bufC, (float*)d_out, (long)NNODES * 2, 1000.f);

  pool_sum_k<<<dim3(NG), 128, 0, stream>>>(h, gf);
  conv(gf, CDIM, 0, CDIM, CDIM, NG, 1.f);
  gemm(wgh1, P(172), nullptr, sg1, NG, 64, CDIM, 64, 0);
  bnk(sg1, nullptr, 173, sg2, NG, 64, 2);
  conv(sg2, 64, 0, 64, 64, NG, 1.f);
  gemm(wgh2, P(178), nullptr, sg3, NG, 2, 64, 2, 0);
  absmax(sg3, NG * 2);
  clampk(sg3, (float*)d_out + (size_t)NNODES * 2, NG * 2, 1000.f);
}
